// TransformerEncoderLayer_14181982012103
// MI455X (gfx1250) — matrix operand order repaired and checked by reading, not compiled, not run on hardware
//
#include <hip/hip_runtime.h>
#include <hip/hip_bf16.h>
#include <math.h>

// ---------------------------------------------------------------------------
// TransformerEncoderLayer for MI455X (gfx1250).
// All matmuls via V_WMMA_F32_16X16X32_BF16 (fp32 accumulate); GEMM tiles are
// staged with GLOBAL_LOAD_ASYNC_TO_LDS_B128 (ASYNCcnt) into double-buffered
// LDS so HBM/L2 traffic overlaps the matrix pipe.
// ---------------------------------------------------------------------------

typedef __attribute__((ext_vector_type(8)))  float  v8f;
typedef __attribute__((ext_vector_type(16))) __bf16 v16bf;
typedef unsigned short u16;
typedef unsigned int   u32;
typedef unsigned long long u64;

#define D_MODEL 1024
#define N_HEADS 16
#define HEAD_DIM 64
#define D_FF 4096
#define SEQ 2048
#define BATCH 2

// ----------------------------- helpers -------------------------------------

__device__ __forceinline__ u16 f2bf(float f) {
  union { float f; u32 u; } c; c.f = f;
  u32 u = c.u;
  u += 0x7FFFu + ((u >> 16) & 1u);   // round-to-nearest-even
  return (u16)(u >> 16);
}

__device__ __forceinline__ float gelu_exact(float v) {
  return 0.5f * v * (1.0f + erff(v * 0.70710678118654752f));
}

// 32-byte fragment: 16 bf16 values per lane, built from two 16B chunks.
union Frag {
  uint4 u[2];
  v16bf v;
};

// Async 16B global -> LDS copy (per-lane), tracked by ASYNCcnt.
__device__ __forceinline__ void async_g2l_b128(u32 lds_addr, u64 gaddr) {
  asm volatile("global_load_async_to_lds_b128 %0, %1, off"
               :: "v"(lds_addr), "v"(gaddr)
               : "memory");
}
__device__ __forceinline__ void wait_async0() {
  asm volatile("s_wait_asynccnt 0" ::: "memory");
}

// ----------------------------- cast kernel ---------------------------------

__global__ __launch_bounds__(256)
void cast_f32_to_bf16(const float* __restrict__ in, u16* __restrict__ out, int n) {
  int i = blockIdx.x * blockDim.x + threadIdx.x;
  if (i < n) out[i] = f2bf(in[i]);
}

// ----------------------------- WMMA GEMM -----------------------------------
// D[M,N] = epilogue( alpha * A[M,K] * B[N,K]^T + bias[N] )
// A, B: bf16 row-major (contraction dim contiguous).
// Block: 256 threads = 8 wave32s (4 in M x 2 in N); wave tile 32x64;
// block tile 128x128, K-step 32. Double-buffered LDS, async global->LDS.

#define BM 128
#define BN 128
#define BK 32
#define LDSS 40   // LDS row stride in halfs (32 + 8 pad), 80B rows keep 16B align

// Issue the async loads for one 128x32 A tile + 128x32 B tile.
// OOB rows are clamped (their results are never stored), so no divergence and
// EXEC stays all-ones for the WMMA pipe.
__device__ __forceinline__ void load_tiles_async(
    const u16* __restrict__ A, int lda, int m0, int M,
    const u16* __restrict__ B, int ldb, int n0, int N,
    int k0, u16* As, u16* Bs, int tid) {
#pragma unroll
  for (int it = 0; it < 2; ++it) {
    int c  = tid + it * 256;        // chunk id 0..511
    int r  = c >> 2;                // tile row 0..127
    int kk = (c & 3) * 8;           // half offset within 32-wide K slice
    int gm = m0 + r; if (gm >= M) gm = M - 1;
    int gn = n0 + r; if (gn >= N) gn = N - 1;
    u32 la = (u32)(u64)(void*)&As[r * LDSS + kk];
    u32 lb = (u32)(u64)(void*)&Bs[r * LDSS + kk];
    u64 ga = (u64)(const void*)(A + (size_t)gm * lda + k0 + kk);
    u64 gb = (u64)(const void*)(B + (size_t)gn * ldb + k0 + kk);
    async_g2l_b128(la, ga);
    async_g2l_b128(lb, gb);
  }
}

__global__ __launch_bounds__(256)
void gemm_bf16_wmma(int M, int N, int K,
                    const u16* __restrict__ A, int lda,
                    const u16* __restrict__ B, int ldb,
                    const float* __restrict__ bias,
                    float* __restrict__ outF,
                    u16*   __restrict__ outB, int ldc,
                    float alpha, int do_gelu, float postscale) {
  __shared__ u16 As[2][BM * LDSS];
  __shared__ u16 Bs[2][BN * LDSS];

  const int tid  = threadIdx.x;
  const int lane = tid & 31;
  const int wave = tid >> 5;
  const int wr   = wave & 3;   // wave row (M dir, 4 waves)
  const int wc   = wave >> 2;  // wave col (N dir, 2 waves)

  const int m0 = blockIdx.y * BM;
  const int n0 = blockIdx.x * BN;

  v8f acc[2][4];
#pragma unroll
  for (int mi = 0; mi < 2; ++mi)
#pragma unroll
    for (int ni = 0; ni < 4; ++ni)
#pragma unroll
      for (int r = 0; r < 8; ++r) acc[mi][ni][r] = 0.0f;

  const int ksel = lane >> 4;       // 0/1 : which K-half of the fragment
  const int mrow = lane & 15;

  // ---- prologue: stage first K tile ----
  load_tiles_async(A, lda, m0, M, B, ldb, n0, N, 0, As[0], Bs[0], tid);
  wait_async0();
  __syncthreads();

  int cur = 0;
  for (int k0 = 0; k0 < K; k0 += BK) {
    // ---- issue async loads for the next tile into the other buffer ----
    if (k0 + BK < K)
      load_tiles_async(A, lda, m0, M, B, ldb, n0, N, k0 + BK,
                       As[cur ^ 1], Bs[cur ^ 1], tid);

    // ---- fragments from resident LDS buffer (ISA 16-bit A/B layouts) ----
    const u16* Ac = As[cur];
    const u16* Bc = Bs[cur];
    Frag af[2];
#pragma unroll
    for (int mi = 0; mi < 2; ++mi) {
      int r = wr * 32 + mi * 16 + mrow;
      af[mi].u[0] = *reinterpret_cast<const uint4*>(&Ac[r * LDSS + ksel * 8]);
      af[mi].u[1] = *reinterpret_cast<const uint4*>(&Ac[r * LDSS + 16 + ksel * 8]);
    }
    Frag bf[4];
    const int koff = ksel * 8;
#pragma unroll
    for (int ni = 0; ni < 4; ++ni) {
      int r = wc * 64 + ni * 16 + mrow;
      bf[ni].u[0] = *reinterpret_cast<const uint4*>(&Bc[r * LDSS + koff]);
      bf[ni].u[1] = *reinterpret_cast<const uint4*>(&Bc[r * LDSS + koff + 16]);
    }

#pragma unroll
    for (int mi = 0; mi < 2; ++mi)
#pragma unroll
      for (int ni = 0; ni < 4; ++ni)
        acc[mi][ni] = __builtin_amdgcn_wmma_f32_16x16x32_bf16(
            false, af[mi].v, false, bf[ni].v, (short)0, acc[mi][ni],
            false, false);

    // next tile's async writes must land; all waves must be done reading
    wait_async0();
    __syncthreads();
    cur ^= 1;
  }

  // ---- epilogue: alpha, bias, gelu, postscale; f32 and/or bf16 stores ----
#pragma unroll
  for (int mi = 0; mi < 2; ++mi) {
    int mbase = m0 + wr * 32 + mi * 16 + ((lane < 16) ? 0 : 8);
#pragma unroll
    for (int ni = 0; ni < 4; ++ni) {
      int col = n0 + wc * 64 + ni * 16 + (lane & 15);
      if (col >= N) continue;
      float bsv = bias ? bias[col] : 0.0f;
#pragma unroll
      for (int r = 0; r < 8; ++r) {
        int row = mbase + r;
        if (row >= M) continue;
        float v = acc[mi][ni][r] * alpha + bsv;
        if (do_gelu) v = gelu_exact(v);
        v *= postscale;
        size_t o = (size_t)row * ldc + col;
        if (outF) outF[o] = v;
        if (outB) outB[o] = f2bf(v);
      }
    }
  }
}

// ------------------------ V transpose (per head, e-major) -------------------
// vT[(b*H+h)*64 + e][s] = qkv[b*S+s][2048 + h*64 + e]   (bf16)
__global__ __launch_bounds__(256)
void transpose_v(const u16* __restrict__ qkv, u16* __restrict__ vT, int total) {
  int i = blockIdx.x * blockDim.x + threadIdx.x;
  if (i >= total) return;
  int s    = i & (SEQ - 1);
  int rest = i >> 11;            // /SEQ
  int e    = rest & (HEAD_DIM - 1);
  int bh   = rest >> 6;          // /64
  int b    = bh >> 4;            // /N_HEADS
  int h    = bh & (N_HEADS - 1);
  vT[i] = qkv[(size_t)(b * SEQ + s) * (3 * D_MODEL) + 2 * D_MODEL + h * HEAD_DIM + e];
}

// -------------------- column softmax (axis = queries, t) --------------------
// One thread per column s; online max/sum over t; emit bf16 probabilities.
__global__ __launch_bounds__(256)
void softmax_col(const float* __restrict__ scores, u16* __restrict__ P, int S) {
  int col = blockIdx.x * blockDim.x + threadIdx.x;
  if (col >= S) return;
  float m = -3.4e38f, sum = 0.0f;
  for (int t = 0; t < S; ++t) {
    float v = scores[(size_t)t * S + col];
    if (v > m) { sum *= __expf(m - v); m = v; }
    sum += __expf(v - m);
  }
  float inv = 1.0f / sum;
  for (int t = 0; t < S; ++t) {
    float v = scores[(size_t)t * S + col];
    P[(size_t)t * S + col] = f2bf(__expf(v - m) * inv);
  }
}

// -------------------- residual add + LayerNorm (wave32 reduce) --------------
__global__ __launch_bounds__(256)
void add_layernorm(const float* __restrict__ a, const float* __restrict__ b,
                   const float* __restrict__ g, const float* __restrict__ be,
                   float* __restrict__ outF, u16* __restrict__ outB, int D) {
  __shared__ float red[16];
  const int tid  = threadIdx.x;
  const int lane = tid & 31;
  const int wave = tid >> 5;
  const size_t base = (size_t)blockIdx.x * D;

  float s = 0.0f, s2 = 0.0f;
  for (int c = tid; c < D; c += 256) {
    float v = a[base + c] + b[base + c];
    s += v; s2 += v * v;
  }
#pragma unroll
  for (int o = 16; o > 0; o >>= 1) {
    s  += __shfl_xor(s,  o, 32);
    s2 += __shfl_xor(s2, o, 32);
  }
  if (lane == 0) { red[wave] = s; red[8 + wave] = s2; }
  __syncthreads();
  if (wave == 0) {
    s  = (lane < 8) ? red[lane]     : 0.0f;
    s2 = (lane < 8) ? red[8 + lane] : 0.0f;
#pragma unroll
    for (int o = 16; o > 0; o >>= 1) {
      s  += __shfl_xor(s,  o, 32);
      s2 += __shfl_xor(s2, o, 32);
    }
    if (lane == 0) { red[0] = s; red[1] = s2; }
  }
  __syncthreads();
  float mu  = red[0] / (float)D;
  float var = red[1] / (float)D - mu * mu;
  float inv = rsqrtf(var + 1e-5f);
  for (int c = tid; c < D; c += 256) {
    float v = a[base + c] + b[base + c];
    float y = (v - mu) * inv * g[c] + be[c];
    if (outF) outF[base + c] = y;
    if (outB) outB[base + c] = f2bf(y);
  }
}

// ----------------------------- host driver ----------------------------------

static inline size_t align_up(size_t x) { return (x + 255) & ~(size_t)255; }

extern "C" void kernel_launch(void* const* d_in, const int* in_sizes, int n_in,
                              void* d_out, int out_size, void* d_ws, size_t ws_size,
                              hipStream_t stream) {
  (void)in_sizes; (void)n_in; (void)out_size; (void)ws_size;

  const float* src   = (const float*)d_in[0];
  const float* w_qkv = (const float*)d_in[1];
  const float* b_qkv = (const float*)d_in[2];
  const float* w_out = (const float*)d_in[3];
  const float* b_out = (const float*)d_in[4];
  const float* w_ff1 = (const float*)d_in[5];
  const float* b_ff1 = (const float*)d_in[6];
  const float* w_ff2 = (const float*)d_in[7];
  const float* b_ff2 = (const float*)d_in[8];
  const float* ln1g  = (const float*)d_in[9];
  const float* ln1b  = (const float*)d_in[10];
  const float* ln2g  = (const float*)d_in[11];
  const float* ln2b  = (const float*)d_in[12];

  const int M  = BATCH * SEQ;          // 4096 token rows
  const int D  = D_MODEL;              // 1024
  const int DF = D_FF;                 // 4096
  const int S  = SEQ;

  // ---- workspace carve-out (bf16 stored as u16) ----
  char* ws = (char*)d_ws;
  size_t cur = 0;
  auto take = [&](size_t bytes) { char* p = ws + cur; cur += align_up(bytes); return p; };

  u16*   srcB   = (u16*)  take((size_t)M * D * 2);            //  8 MB
  u16*   wqkvB  = (u16*)  take((size_t)3 * D * D * 2);        //  6 MB
  u16*   woutB  = (u16*)  take((size_t)D * D * 2);            //  2 MB
  u16*   wff1B  = (u16*)  take((size_t)DF * D * 2);           //  8 MB
  u16*   wff2B  = (u16*)  take((size_t)D * DF * 2);           //  8 MB
  u16*   qkvB   = (u16*)  take((size_t)M * 3 * D * 2);        // 24 MB
  u16*   vT     = (u16*)  take((size_t)M * D * 2);            //  8 MB
  float* scores = (float*)take((size_t)S * S * 4);            // 16 MB
  u16*   probs  = (u16*)  take((size_t)S * S * 2);            //  8 MB
  u16*   attnB  = (u16*)  take((size_t)M * D * 2);            //  8 MB
  float* attnO  = (float*)take((size_t)M * D * 4);            // 16 MB
  float* x1     = (float*)take((size_t)M * D * 4);            // 16 MB
  u16*   x1B    = (u16*)  take((size_t)M * D * 2);            //  8 MB
  u16*   hB     = (u16*)  take((size_t)M * DF * 2);           // 32 MB
  float* ff     = (float*)take((size_t)M * D * 4);            // 16 MB

  const dim3 blk(256);

  // ---- 1. fp32 -> bf16 operand casts ----
  {
    int n;
    n = M * D;      cast_f32_to_bf16<<<(n + 255) / 256, blk, 0, stream>>>(src,   srcB,  n);
    n = 3 * D * D;  cast_f32_to_bf16<<<(n + 255) / 256, blk, 0, stream>>>(w_qkv, wqkvB, n);
    n = D * D;      cast_f32_to_bf16<<<(n + 255) / 256, blk, 0, stream>>>(w_out, woutB, n);
    n = DF * D;     cast_f32_to_bf16<<<(n + 255) / 256, blk, 0, stream>>>(w_ff1, wff1B, n);
    n = D * DF;     cast_f32_to_bf16<<<(n + 255) / 256, blk, 0, stream>>>(w_ff2, wff2B, n);
  }

  // ---- 2. QKV projection: qkv = src @ Wqkv^T + b  (bf16 out, ld = 3072) ----
  {
    dim3 grid((3 * D + BN - 1) / BN, (M + BM - 1) / BM);
    gemm_bf16_wmma<<<grid, blk, 0, stream>>>(M, 3 * D, D, srcB, D, wqkvB, D,
                                             b_qkv, nullptr, qkvB, 3 * D,
                                             1.0f, 0, 1.0f);
  }

  // ---- 3. V transpose so PV contraction dim (s) is contiguous ----
  {
    int n = M * D;
    transpose_v<<<(n + 255) / 256, blk, 0, stream>>>(qkvB, vT, n);
  }

  // ---- 4. attention per (b, h): scores -> column softmax -> P @ V ----
  const float sm_scale  = 1.0f / sqrtf((float)HEAD_DIM);   // 0.125
  const float postscale = 1.0f / sqrtf((float)S);
  for (int b = 0; b < BATCH; ++b) {
    for (int h = 0; h < N_HEADS; ++h) {
      const u16* qh = qkvB + (size_t)b * S * 3 * D + h * HEAD_DIM;
      const u16* kh = qkvB + (size_t)b * S * 3 * D + D + h * HEAD_DIM;
      // scores[t,s] = (q_t . k_s) * sm_scale   (S x S, K = 64)
      {
        dim3 grid((S + BN - 1) / BN, (S + BM - 1) / BM);
        gemm_bf16_wmma<<<grid, blk, 0, stream>>>(S, S, HEAD_DIM, qh, 3 * D, kh, 3 * D,
                                                 nullptr, scores, nullptr, S,
                                                 sm_scale, 0, 1.0f);
      }
      // softmax over t (queries) per column s
      softmax_col<<<(S + 255) / 256, blk, 0, stream>>>(scores, probs, S);
      // post[t,e] = gelu(P @ V) / sqrt(S), bf16 into concat activation
      {
        const u16* vh  = vT + (size_t)(b * N_HEADS + h) * HEAD_DIM * S;
        u16* outh = attnB + (size_t)b * S * D + h * HEAD_DIM;
        dim3 grid((HEAD_DIM + BN - 1) / BN, (S + BM - 1) / BM);
        gemm_bf16_wmma<<<grid, blk, 0, stream>>>(S, HEAD_DIM, S, probs, S, vh, S,
                                                 nullptr, nullptr, outh, D,
                                                 1.0f, 1, postscale);
      }
    }
  }

  // ---- 5. output projection ----
  {
    dim3 grid((D + BN - 1) / BN, (M + BM - 1) / BM);
    gemm_bf16_wmma<<<grid, blk, 0, stream>>>(M, D, D, attnB, D, woutB, D,
                                             b_out, attnO, nullptr, D,
                                             1.0f, 0, 1.0f);
  }

  // ---- 6. x = LN(src + attn_out) ----
  add_layernorm<<<M, blk, 0, stream>>>(src, attnO, ln1g, ln1b, x1, x1B, D);

  // ---- 7. FFN up: h = gelu(x @ W1^T + b1), bf16 ----
  {
    dim3 grid((DF + BN - 1) / BN, (M + BM - 1) / BM);
    gemm_bf16_wmma<<<grid, blk, 0, stream>>>(M, DF, D, x1B, D, wff1B, D,
                                             b_ff1, nullptr, hB, DF,
                                             1.0f, 1, 1.0f);
  }

  // ---- 8. FFN down: ff = h @ W2^T + b2 ----
  {
    dim3 grid((D + BN - 1) / BN, (M + BM - 1) / BM);
    gemm_bf16_wmma<<<grid, blk, 0, stream>>>(M, D, DF, hB, DF, wff2B, DF,
                                             b_ff2, ff, nullptr, D,
                                             1.0f, 0, 1.0f);
  }

  // ---- 9. out = LN(x + ff) ----
  add_layernorm<<<M, blk, 0, stream>>>(x1, ff, ln2g, ln2b, (float*)d_out, nullptr, D);
}
